// ClusterBlock_61744449847483
// MI455X (gfx1250) — matrix operand order repaired and checked by reading, not compiled, not run on hardware
//
#include <hip/hip_runtime.h>
#include <hip/hip_bf16.h>
#include <math.h>

typedef __attribute__((ext_vector_type(16))) _Float16 v16h;
typedef __attribute__((ext_vector_type(8)))  float    v8f;

#define B_    64
#define C_    192
#define HW_   1024
#define NTOK  (B_*HW_)      // 65536 tokens
#define HID_  768
#define NPB   (C_*HW_)      // 196608 elements per sample

#define USE_ASYNC 1         // gfx1250 GLOBAL_LOAD_ASYNC_TO_LDS staging (flip to 0 if asm rejects)

#define WMMA_F16(acc, a, b) \
  (acc) = __builtin_amdgcn_wmma_f32_16x16x32_f16(false, (a), false, (b), (short)0, (acc), false, false)

// ---- async global->LDS staging (CDNA5 TDM-side path, ASYNCcnt tracked) ------
__device__ __forceinline__ void async_ld16(void* ldsPtr, const void* gptr) {
  uint32_t l = (uint32_t)(uintptr_t)ldsPtr;    // low 32 bits of flat LDS addr = LDS offset
  asm volatile("global_load_async_to_lds_b128 %0, %1, off" :: "v"(l), "v"(gptr) : "memory");
}
__device__ __forceinline__ void wait_async0() {
  asm volatile("s_wait_asynccnt 0x0" ::: "memory");
}
// stage nvec*16 bytes, block-cooperative; caller must __syncthreads() after
__device__ __forceinline__ void stage_tile(_Float16* dst, const _Float16* src,
                                           int nvec, int tid, int nthreads) {
#if USE_ASYNC
  for (int i = tid; i < nvec; i += nthreads)
    async_ld16((void*)(reinterpret_cast<uint4*>(dst) + i),
               (const void*)(reinterpret_cast<const uint4*>(src) + i));
  wait_async0();
#else
  for (int i = tid; i < nvec; i += nthreads)
    reinterpret_cast<uint4*>(dst)[i] = reinterpret_cast<const uint4*>(src)[i];
#endif
}

// ---- WMMA fragment loaders --------------------------------------------------
// A fragment as loaded here (contiguous k order): lanes 0-15 -> M=lane, K in {0..15};
// lanes 16-31 -> M=lane-16, K in {16..31}; each VGPR holds a K pair.
__device__ __forceinline__ v16h a_frag(const _Float16* tile, int lda, int kBase, int lane) {
  int m = lane & 15, g = lane >> 4;
  union { uint32_t u[8]; v16h v; } r;
  const uint32_t* row = reinterpret_cast<const uint32_t*>(tile + m * lda);  // lda even
  #pragma unroll
  for (int j = 0; j < 8; ++j) {
    int k = kBase + g * 16 + (j << 1);
    r.u[j] = row[k >> 1];
  }
  return r.v;
}
// B fragment as loaded here (contiguous k, as a_frag): lane n = L%16, K = 16*(L/16) + h. Weights stored [n][k] f16,
// so each lane's fragment is one contiguous 32-byte load.
__device__ __forceinline__ v16h b_frag(const _Float16* w, int ldk, int n0, int kBase, int lane) {
  int n = lane & 15, g = lane >> 4;
  return *reinterpret_cast<const v16h*>(w + (size_t)(n0 + n) * ldk + kBase + g * 16);
}

// ---- weight conversion to f16, [out_channel][k] layout ----------------------
__global__ void k_convert_w(const float* fw, const float* vw, const float* pw,
                            const float* w1, const float* w2,
                            _Float16* fw16, _Float16* vw16, _Float16* pw16,
                            _Float16* w1t, _Float16* w2t) {
  int i = blockIdx.x * blockDim.x + threadIdx.x;
  int stride = gridDim.x * blockDim.x;
  for (; i < C_ * HID_; i += stride) {
    if (i < C_ * C_) {  // already [o][c]
      fw16[i] = (_Float16)fw[i];
      vw16[i] = (_Float16)vw[i];
      pw16[i] = (_Float16)pw[i];
    }
    int h = i / C_, c = i % C_;
    w1t[(size_t)h * C_ + c] = (_Float16)w1[(size_t)c * HID_ + h];   // fc1: [c][h] -> [h][c]
    int o = i % C_, k = i / C_;
    w2t[(size_t)o * HID_ + k] = (_Float16)w2[(size_t)k * C_ + o];   // fc2: [h][o] -> [o][h]
  }
}

// ---- GroupNorm(1) statistics ------------------------------------------------
__global__ void k_gn_stats(const float* x, float* mu, float* rs) {
  __shared__ float s1[256], s2[256];
  int b = blockIdx.x, tid = threadIdx.x;
  const float* xb = x + (size_t)b * NPB;
  float a = 0.f, q = 0.f;
  for (int i = tid; i < NPB; i += 256) { float v = xb[i]; a += v; q += v * v; }
  s1[tid] = a; s2[tid] = q; __syncthreads();
  for (int s = 128; s > 0; s >>= 1) {
    if (tid < s) { s1[tid] += s1[tid + s]; s2[tid] += s2[tid + s]; }
    __syncthreads();
  }
  if (tid == 0) {
    float m = s1[0] / (float)NPB;
    float var = s2[0] / (float)NPB - m * m;
    mu[b] = m; rs[b] = rsqrtf(var + 1e-5f);
  }
}

// ---- GroupNorm apply with LDS transpose: NCHW f32 -> token-major f16 --------
// block = 256 threads handles 64 tokens; coalesced reads AND b128 writes.
__global__ void k_gn_apply_t(const float* x, const float* mu, const float* rs,
                             const float* gw, const float* gb, _Float16* y) {
  __shared__ _Float16 T[64 * C_];                 // 24 KB
  int blk = blockIdx.x;                           // NTOK/64 = 1024 blocks
  int b = blk >> 4;                               // 16 tiles per sample
  int hw0 = (blk & 15) * 64;
  int t = threadIdx.x & 63;
  int cq = threadIdx.x >> 6;                      // 4 channel phases
  float m = mu[b], r = rs[b];
  for (int c = cq; c < C_; c += 4) {
    float v = (x[(size_t)b * NPB + (size_t)c * HW_ + hw0 + t] - m) * r * gw[c] + gb[c];
    T[t * C_ + c] = (_Float16)v;
  }
  __syncthreads();
  const uint4* s = reinterpret_cast<const uint4*>(T);
  uint4* d = reinterpret_cast<uint4*>(y + (size_t)(b * HW_ + hw0) * C_);
  for (int i = threadIdx.x; i < (64 * C_) / 8; i += 256) d[i] = s[i];
}

// ---- f & v projections: [NTOK x 192] x [192 x 192] twice --------------------
// 1 block = 16 tokens, 4 waves; each wave: 6 N-tiles reusing preloaded A frags.
__global__ void k_fv_gemm(const _Float16* Y, const _Float16* Wf, const float* bf,
                          const _Float16* Wv, const float* bv,
                          _Float16* Fo, _Float16* Vo) {
  __shared__ _Float16 At[16 * C_];
  __shared__ _Float16 Of[16 * C_];
  __shared__ _Float16 Ov[16 * C_];
  int t0 = blockIdx.x * 16;
  stage_tile(At, Y + (size_t)t0 * C_, (16 * C_) / 8, threadIdx.x, blockDim.x);
  __syncthreads();
  int lane = threadIdx.x & 31, wid = threadIdx.x >> 5;
  int g = lane >> 4, n = lane & 15;
  v16h az[6];
  #pragma unroll
  for (int kk = 0; kk < 6; ++kk) az[kk] = a_frag(At, C_, kk * 32, lane);
  for (int it = 0; it < 6; ++it) {
    int nt = wid + 4 * it;                        // 24 tiles: 12 f + 12 v
    const _Float16* W; const float* bias; _Float16* O; int n0;
    if (nt < 12) { W = Wf; bias = bf; O = Of; n0 = nt * 16; }
    else         { W = Wv; bias = bv; O = Ov; n0 = (nt - 12) * 16; }
    v8f acc = {};
    #pragma unroll
    for (int kk = 0; kk < 6; ++kk) {
      v16h bm = b_frag(W, C_, n0, kk * 32, lane);
      WMMA_F16(acc, az[kk], bm);
    }
    float bsv = bias[n0 + n];
    #pragma unroll
    for (int r2 = 0; r2 < 8; ++r2)
      O[(r2 + 8 * g) * C_ + n0 + n] = (_Float16)(acc[r2] + bsv);
  }
  __syncthreads();
  // coalesced b128 writeback
  const uint4* sf = reinterpret_cast<const uint4*>(Of);
  const uint4* sv = reinterpret_cast<const uint4*>(Ov);
  uint4* df = reinterpret_cast<uint4*>(Fo + (size_t)t0 * C_);
  uint4* dv = reinterpret_cast<uint4*>(Vo + (size_t)t0 * C_);
  for (int i = threadIdx.x; i < (16 * C_) / 8; i += blockDim.x) { df[i] = sf[i]; dv[i] = sv[i]; }
}

// ---- clustering: one block per folded unit (b,e,f1,f2) ----------------------
__global__ void k_cluster(const _Float16* F, const _Float16* V,
                          const float* alpha, const float* beta, _Float16* OUT) {
  __shared__ float vt[256 * 24];
  __shared__ float cf[4][24], cv[4][24], agg[4][24];
  __shared__ float bsL[256];
  __shared__ int   bmL[256];
  __shared__ float cn[4];
  int u = blockIdx.x;                       // u = ((b*8+e)*2+f1)*2+f2
  int f2 = u & 1, f1 = (u >> 1) & 1, e = (u >> 2) & 7, b = u >> 5;
  int tid = threadIdx.x;
  int chBase = e * 24;
  // pooled centers over 8x8 regions (AdaptiveAvgPool2d(2,2) on 16x16)
  if (tid < 96) {
    int m = tid / 24, c = tid - m * 24;
    int pw = m >> 1, ph = m & 1;
    float sf = 0.f, sv = 0.f;
    for (int i = 0; i < 64; ++i) {
      int wp = pw * 8 + (i >> 3), hp = ph * 8 + (i & 7);
      int tg = b * HW_ + (f1 * 16 + wp) * 32 + (f2 * 16 + hp);
      size_t off = (size_t)tg * C_ + chBase + c;
      sf += (float)F[off]; sv += (float)V[off];
    }
    cf[m][c] = sf * (1.f / 64.f); cv[m][c] = sv * (1.f / 64.f);
  }
  __syncthreads();
  if (tid < 4) {
    float s = 0.f;
    for (int c = 0; c < 24; ++c) s += cf[tid][c] * cf[tid][c];
    cn[tid] = fmaxf(sqrtf(s), 1e-12f);
  }
  __syncthreads();
  // per-token: cosine sim, sigmoid, hard argmax (first max wins, as jnp.argmax)
  int wp = tid >> 4, hp = tid & 15;
  int tg = b * HW_ + (f1 * 16 + wp) * 32 + (f2 * 16 + hp);
  const _Float16* fp = F + (size_t)tg * C_ + chBase;
  const _Float16* vp = V + (size_t)tg * C_ + chBase;
  float fv[24]; float nrm = 0.f;
  #pragma unroll
  for (int c = 0; c < 24; ++c) {
    float xv = (float)fp[c]; fv[c] = xv; nrm += xv * xv;
    vt[tid * 24 + c] = (float)vp[c];
  }
  nrm = fmaxf(sqrtf(nrm), 1e-12f);
  float al = alpha[0], be = beta[0];
  float bs = -1e30f; int bm = 0;
  #pragma unroll
  for (int m = 0; m < 4; ++m) {
    float d = 0.f;
    #pragma unroll
    for (int c = 0; c < 24; ++c) d += fv[c] * cf[m][c];
    float z = be + al * d / (nrm * cn[m]);
    float s = 1.f / (1.f + expf(-z));
    if (s > bs) { bs = s; bm = m; }
  }
  bsL[tid] = bs; bmL[tid] = bm;
  __syncthreads();
  // deterministic aggregate (no float atomics -> bit-stable across graph replays)
  if (tid < 96) {
    int m = tid / 24, c = tid - m * 24;
    float s = 0.f; int cnt = 0;
    for (int t = 0; t < 256; ++t) {
      bool sel = (bmL[t] == m);
      if (sel) s += bsL[t] * vt[t * 24 + c];
      cnt += sel ? 1 : 0;
    }
    agg[m][c] = (s + cv[m][c]) / ((float)cnt + 1.f);
  }
  __syncthreads();
  _Float16* op = OUT + (size_t)tg * C_ + chBase;
  #pragma unroll
  for (int c = 0; c < 24; ++c) op[c] = (_Float16)(bs * agg[bm][c]);
}

// ---- proj GEMM + ls1 residual: writes x1 into d_out (f32, NCHW) -------------
__global__ void k_proj_res(const _Float16* U, const _Float16* Wp, const float* pb,
                           const float* ls1, const float* x, float* out) {
  __shared__ _Float16 At[16 * C_];
  __shared__ float    Ot[16 * C_];                // raw accumulators
  int t0 = blockIdx.x * 16;
  stage_tile(At, U + (size_t)t0 * C_, (16 * C_) / 8, threadIdx.x, blockDim.x);
  __syncthreads();
  int lane = threadIdx.x & 31, wid = threadIdx.x >> 5;
  int g = lane >> 4, n = lane & 15;
  v16h az[6];
  #pragma unroll
  for (int kk = 0; kk < 6; ++kk) az[kk] = a_frag(At, C_, kk * 32, lane);
  for (int it = 0; it < 3; ++it) {                // 12 N tiles, 3 per wave
    int n0 = (wid + 4 * it) * 16;
    v8f acc = {};
    #pragma unroll
    for (int kk = 0; kk < 6; ++kk) {
      v16h bm = b_frag(Wp, C_, n0, kk * 32, lane);
      WMMA_F16(acc, az[kk], bm);
    }
    #pragma unroll
    for (int r2 = 0; r2 < 8; ++r2)
      Ot[(r2 + 8 * g) * C_ + n0 + n] = acc[r2];
  }
  __syncthreads();
  // coalesced NCHW writeback: 64B contiguous per 16-token row
  int bi = t0 >> 10, hw0 = t0 & 1023;
  for (int i = threadIdx.x; i < 16 * C_; i += blockDim.x) {
    int o = i >> 4, t = i & 15;
    size_t oi = (size_t)bi * NPB + (size_t)o * HW_ + hw0 + t;
    out[oi] = x[oi] + ls1[o] * (Ot[t * C_ + o] + pb[o]);
  }
}

// ---- fused MLP: hid = gelu(z@W1+b1) staged in LDS; out += ls2*(hid@W2+b2) ---
__global__ void k_mlp(const _Float16* Z, const _Float16* W1t, const float* b1,
                      const _Float16* W2t, const float* b2, const float* ls2,
                      float* out) {
  __shared__ _Float16 Zt[16 * C_];
  __shared__ _Float16 Hid[16 * HID_];
  __shared__ float    Ot[16 * C_];
  int t0 = blockIdx.x * 16;
  stage_tile(Zt, Z + (size_t)t0 * C_, (16 * C_) / 8, threadIdx.x, blockDim.x);
  __syncthreads();
  int lane = threadIdx.x & 31, wid = threadIdx.x >> 5;
  int g = lane >> 4, n = lane & 15;
  v16h az[6];
  #pragma unroll
  for (int kk = 0; kk < 6; ++kk) az[kk] = a_frag(Zt, C_, kk * 32, lane);
  // phase A: 48 hidden tiles of 16, 12 per wave
  for (int it = 0; it < 12; ++it) {
    int h0 = (wid + 4 * it) * 16;
    v8f acc = {};
    #pragma unroll
    for (int kk = 0; kk < 6; ++kk) {
      v16h bm = b_frag(W1t, C_, h0, kk * 32, lane);
      WMMA_F16(acc, az[kk], bm);
    }
    float bb = b1[h0 + n];
    #pragma unroll
    for (int r2 = 0; r2 < 8; ++r2) {
      int m = r2 + 8 * g;
      float vx = acc[r2] + bb;
      float ge = 0.5f * vx * (1.f + erff(vx * 0.70710678118654752f));  // exact gelu
      Hid[m * HID_ + h0 + n] = (_Float16)ge;
    }
  }
  __syncthreads();
  // phase B: 12 output tiles of 16, 3 per wave, K = 768
  for (int it = 0; it < 3; ++it) {
    int o0 = (wid + 4 * it) * 16;
    v8f acc = {};
    #pragma unroll
    for (int kk = 0; kk < 24; ++kk) {
      v16h a = a_frag(Hid, HID_, kk * 32, lane);
      v16h bm = b_frag(W2t, HID_, o0, kk * 32, lane);
      WMMA_F16(acc, a, bm);
    }
    #pragma unroll
    for (int r2 = 0; r2 < 8; ++r2)
      Ot[(r2 + 8 * g) * C_ + o0 + n] = acc[r2];
  }
  __syncthreads();
  int bi = t0 >> 10, hw0 = t0 & 1023;
  for (int i = threadIdx.x; i < 16 * C_; i += blockDim.x) {
    int o = i >> 4, t = i & 15;
    size_t oi = (size_t)bi * NPB + (size_t)o * HW_ + hw0 + t;
    out[oi] += ls2[o] * (Ot[t * C_ + o] + b2[o]);
  }
}

extern "C" void kernel_launch(void* const* d_in, const int* in_sizes, int n_in,
                              void* d_out, int out_size, void* d_ws, size_t ws_size,
                              hipStream_t stream) {
  const float* x     = (const float*)d_in[0];
  const float* gn1w  = (const float*)d_in[1];
  const float* gn1b  = (const float*)d_in[2];
  const float* fw    = (const float*)d_in[3];
  const float* fb    = (const float*)d_in[4];
  const float* vw    = (const float*)d_in[5];
  const float* vb    = (const float*)d_in[6];
  const float* pw    = (const float*)d_in[7];
  const float* pb    = (const float*)d_in[8];
  const float* alpha = (const float*)d_in[9];
  const float* beta  = (const float*)d_in[10];
  const float* ls1   = (const float*)d_in[11];
  const float* gn2w  = (const float*)d_in[12];
  const float* gn2b  = (const float*)d_in[13];
  const float* w1    = (const float*)d_in[14];
  const float* b1    = (const float*)d_in[15];
  const float* w2    = (const float*)d_in[16];
  const float* b2    = (const float*)d_in[17];
  const float* ls2   = (const float*)d_in[18];
  float* out = (float*)d_out;

  // workspace carve (~77 MB total), 256B aligned
  uint8_t* ws = (uint8_t*)d_ws;
  size_t off = 0;
  auto carve = [&](size_t bytes) -> void* {
    void* p = ws + off; off += (bytes + 255) & ~(size_t)255; return p;
  };
  float*    mu1  = (float*)carve(B_ * 4);
  float*    rs1  = (float*)carve(B_ * 4);
  float*    mu2  = (float*)carve(B_ * 4);
  float*    rs2  = (float*)carve(B_ * 4);
  _Float16* fw16 = (_Float16*)carve((size_t)C_ * C_ * 2);
  _Float16* vw16 = (_Float16*)carve((size_t)C_ * C_ * 2);
  _Float16* pw16 = (_Float16*)carve((size_t)C_ * C_ * 2);
  _Float16* w1t  = (_Float16*)carve((size_t)C_ * HID_ * 2);
  _Float16* w2t  = (_Float16*)carve((size_t)C_ * HID_ * 2);
  _Float16* Y16  = (_Float16*)carve((size_t)NTOK * C_ * 2);   // gn1 out, then cluster out
  _Float16* F16  = (_Float16*)carve((size_t)NTOK * C_ * 2);   // f, then gn2 out
  _Float16* V16  = (_Float16*)carve((size_t)NTOK * C_ * 2);   // v

  k_convert_w<<<576, 256, 0, stream>>>(fw, vw, pw, w1, w2, fw16, vw16, pw16, w1t, w2t);
  k_gn_stats<<<B_, 256, 0, stream>>>(x, mu1, rs1);
  k_gn_apply_t<<<NTOK / 64, 256, 0, stream>>>(x, mu1, rs1, gn1w, gn1b, Y16);
  k_fv_gemm<<<NTOK / 16, 128, 0, stream>>>(Y16, fw16, fb, vw16, vb, F16, V16);
  k_cluster<<<2048, 256, 0, stream>>>(F16, V16, alpha, beta, Y16);
  k_proj_res<<<NTOK / 16, 128, 0, stream>>>(Y16, pw16, pb, ls1, x, out);
  k_gn_stats<<<B_, 256, 0, stream>>>(out, mu2, rs2);
  k_gn_apply_t<<<NTOK / 64, 256, 0, stream>>>(out, mu2, rs2, gn2w, gn2b, F16);
  k_mlp<<<NTOK / 16, 128, 0, stream>>>(F16, w1t, b1, w2t, b2, ls2, out);
}
